// CombinedLSTMWithStatic2Hop_32229434589314
// MI455X (gfx1250) — matrix operand order repaired and checked by reading, not compiled, not run on hardware
//
#include <hip/hip_runtime.h>
#include <hip/hip_bf16.h>

#define BB 4
#define TT 96
#define NN 2000
#define BN 8000
#define FD 8
#define FS 16
#define HH 128
#define GG 512
#define EE 16000
#define AROW 168          // LDS row stride (bf16) for LSTM A buffer, mult of 8, bank-friendly
#define GROW 264          // LDS row stride (bf16) for GEMM A buffer

typedef __attribute__((ext_vector_type(16))) __bf16 v16bf;
typedef __attribute__((ext_vector_type(8)))  float  v8f;

union Frag16 { v16bf v; uint4 q[2]; };

#define LOG2E 1.44269504088896340736f

// Branch-free fast sigmoid/tanh on raw v_exp_f32 / v_rcp_f32.
// Saturation is exact: exp2 -> 0/inf gives sig -> 0/1, tanh -> -1/+1.
__device__ __forceinline__ float sig_(float x) {
    return __builtin_amdgcn_rcpf(1.0f + __builtin_amdgcn_exp2f(-LOG2E * x));
}
__device__ __forceinline__ float tanh_(float x) {
    const float e = __builtin_amdgcn_exp2f((2.0f * LOG2E) * x);
    return 1.0f - 2.0f * __builtin_amdgcn_rcpf(e + 1.0f);
}

// ---------------------------------------------------------------------------
// Kernel 1: LSTM over T steps for a 16-row tile + static encoder epilogue.
// 8 waves / block. Wave w owns gate columns [16w,16w+16) of all 4 gates
// (B n-tiles w, w+8, w+16, w+24), so c/h state stays in WMMA C-layout regs.
// Extended K = 160: [h(128) | x(8) | pad(24)]. Weights resident in VGPRs.
// ---------------------------------------------------------------------------
__global__ __launch_bounds__(256, 1)
void k_lstm_static(const float* __restrict__ dyn,    // [B,T,N,FD]
                   const float* __restrict__ sta,    // [BN,FS]
                   const float* __restrict__ W_ih,   // [GG,FD]
                   const float* __restrict__ W_hh,   // [GG,HH]
                   const float* __restrict__ b_ih,   // [GG]
                   const float* __restrict__ b_hh,   // [GG]
                   const float* __restrict__ W_sta,  // [HH,FS]
                   const float* __restrict__ b_sta,  // [HH]
                   float* __restrict__ z_dyn,        // [BN,HH]
                   float* __restrict__ s_out)        // [BN,HH]
{
    __shared__ __align__(16) __bf16 Alds[16 * AROW];

    const int tid  = threadIdx.x;
    const int lane = tid & 31;
    const int w    = tid >> 5;
    const int half = lane >> 4;
    const int lrow = lane & 15;
    const int col  = (w << 4) | lrow;     // owned h-column (N = lane%16)
    const int r0   = blockIdx.x * 16;     // 16 | NN, tile never spans batches
    const int b    = r0 / NN;
    const int n0   = r0 % NN;

    // ---- B fragments (extended weights) resident in registers ----
    v16bf bfrag[4][5];
#pragma unroll
    for (int j = 0; j < 4; ++j) {
        const int ncol = ((w + 8 * j) << 4) | lrow;     // gate j, column block
#pragma unroll
        for (int kc = 0; kc < 5; ++kc) {
            v16bf f;
#pragma unroll
            for (int e = 0; e < 16; ++e) {
                const int K = kc * 32 + (half << 4) + e;   // B: K = 16*half + e
                float v = 0.0f;
                if (K < HH)           v = W_hh[ncol * HH + K];
                else if (K < HH + FD) v = W_ih[ncol * FD + (K - HH)];
                f[e] = (__bf16)v;
            }
            bfrag[j][kc] = f;
        }
    }

    // per-lane gate biases (column `col` of each gate)
    const float bi = b_ih[col]          + b_hh[col];
    const float bf = b_ih[HH + col]     + b_hh[HH + col];
    const float bg = b_ih[2 * HH + col] + b_hh[2 * HH + col];
    const float bo = b_ih[3 * HH + col] + b_hh[3 * HH + col];

    // ---- init A buffer: h=0, pad=0, then load x_0 ----
    for (int i = tid; i < 16 * AROW; i += 256) Alds[i] = (__bf16)0.0f;
    __syncthreads();
    if (w < 4) {
        const int idx = (w << 5) | lane;          // 0..127
        const int row = idx >> 3, f = idx & 7;
        const float xv = dyn[(((size_t)b * TT + 0) * NN + (n0 + row)) * FD + f];
        Alds[row * AROW + HH + f] = (__bf16)xv;
    }

    float c[8], h[8];
#pragma unroll
    for (int e = 0; e < 8; ++e) { c[e] = 0.0f; h[e] = 0.0f; }

    for (int t = 0; t < TT; ++t) {
        __syncthreads();                          // h_{t-1}, x_t visible
        Frag16 a[5];
#pragma unroll
        for (int kc = 0; kc < 5; ++kc) {          // A: K = 16*half + e
            const int base = lrow * AROW + kc * 32 + (half << 4);
            a[kc].q[0] = *(const uint4*)&Alds[base];
            a[kc].q[1] = *(const uint4*)&Alds[base + 8];
        }
        v8f acc[4];
#pragma unroll
        for (int j = 0; j < 4; ++j) {
            v8f z = {0.f, 0.f, 0.f, 0.f, 0.f, 0.f, 0.f, 0.f};
#pragma unroll
            for (int kc = 0; kc < 5; ++kc)
                z = __builtin_amdgcn_wmma_f32_16x16x32_bf16(
                        false, a[kc].v, false, bfrag[j][kc], (short)0, z, false, false);
            acc[j] = z;
        }
        __syncthreads();                          // all A reads complete
#pragma unroll
        for (int e = 0; e < 8; ++e) {
            const float ig = sig_(acc[0][e] + bi);
            const float fg = sig_(acc[1][e] + bf);
            const float gg = tanh_(acc[2][e] + bg);
            const float og = sig_(acc[3][e] + bo);
            c[e] = fg * c[e] + ig * gg;
            h[e] = og * tanh_(c[e]);
            const int M = e + (half << 3);
            Alds[M * AROW + col] = (__bf16)h[e];  // next step's A operand
        }
        if (t + 1 < TT && w < 4) {
            const int idx = (w << 5) | lane;
            const int row = idx >> 3, f = idx & 7;
            const float xv = dyn[(((size_t)b * TT + (t + 1)) * NN + (n0 + row)) * FD + f];
            Alds[row * AROW + HH + f] = (__bf16)xv;
        }
    }

    // ---- epilogue: z_dyn (f32 h) + static encoder s_tilde ----
#pragma unroll
    for (int e = 0; e < 8; ++e) {
        const int M = e + (half << 3);
        const int gr = r0 + M;
        z_dyn[(size_t)gr * HH + col] = h[e];
        float s = b_sta[col];
#pragma unroll
        for (int k = 0; k < FS; ++k)
            s += sta[(size_t)gr * FS + k] * W_sta[col * FS + k];
        s_out[(size_t)gr * HH + col] = fmaxf(s, 0.0f);
    }
}

// ---------------------------------------------------------------------------
// Kernel 2: out[.,128] = act( [X0*scale | X1] (K=256) @ [W0^T ; W1^T] + bias )
// 64 rows / block, 8 waves; wave w owns n-tile w with B-fragments in regs.
// scale = 1/max(cnt,1) when cnt != nullptr (SAGE mean aggregation).
// ---------------------------------------------------------------------------
__global__ __launch_bounds__(256, 1)
void k_gemm256(const float* __restrict__ X0, const float* __restrict__ X1,
               const float* __restrict__ cnt,
               const float* __restrict__ W0, int ld0,
               const float* __restrict__ W1, int ld1,
               const float* __restrict__ bias, int relu,
               float* __restrict__ out)
{
    __shared__ __align__(16) __bf16 Alds[64 * GROW];

    const int tid  = threadIdx.x;
    const int lane = tid & 31;
    const int w    = tid >> 5;
    const int half = lane >> 4;
    const int lrow = lane & 15;
    const int ncol = (w << 4) | lrow;
    const int r0   = blockIdx.x * 64;     // 125 blocks * 64 = 8000 exactly

    // B fragments for this wave's n-tile (K = 0..255 -> 8 chunks), in regs
    v16bf bfrag[8];
#pragma unroll
    for (int kc = 0; kc < 8; ++kc) {
        v16bf f;
#pragma unroll
        for (int e = 0; e < 16; ++e) {
            const int K = kc * 32 + (half << 4) + e;
            const float v = (K < 128) ? W0[ncol * ld0 + K] : W1[ncol * ld1 + (K - 128)];
            f[e] = (__bf16)v;
        }
        bfrag[kc] = f;
    }
    const float bv = bias[ncol];

    // Stage A = [X0*scale | X1] in LDS (bf16)
    for (int idx = tid; idx < 64 * 256; idx += 256) {
        const int row = idx >> 8;
        const int cc  = idx & 255;
        const int gr  = r0 + row;
        float v;
        if (cc < 128) {
            v = X0[(size_t)gr * HH + cc];
            if (cnt) v *= __builtin_amdgcn_rcpf(fmaxf(cnt[gr], 1.0f));
        } else {
            v = X1[(size_t)gr * HH + (cc - 128)];
        }
        Alds[row * GROW + cc] = (__bf16)v;
    }
    __syncthreads();

#pragma unroll
    for (int rt = 0; rt < 4; ++rt) {
        v8f z = {0.f, 0.f, 0.f, 0.f, 0.f, 0.f, 0.f, 0.f};
#pragma unroll
        for (int kc = 0; kc < 8; ++kc) {
            Frag16 a;
            const int base = (rt * 16 + lrow) * GROW + kc * 32 + (half << 4);
            a.q[0] = *(const uint4*)&Alds[base];
            a.q[1] = *(const uint4*)&Alds[base + 8];
            z = __builtin_amdgcn_wmma_f32_16x16x32_bf16(
                    false, a.v, false, bfrag[kc], (short)0, z, false, false);
        }
#pragma unroll
        for (int e = 0; e < 8; ++e) {
            const int gr = r0 + rt * 16 + e + (half << 3);
            float v = z[e] + bv;
            if (relu) v = fmaxf(v, 0.0f);
            out[(size_t)gr * HH + ncol] = v;
        }
    }
}

// ---------------------------------------------------------------------------
// SAGE edge aggregation: 32 lanes per edge, 4 floats per lane, float atomics.
// ---------------------------------------------------------------------------
__global__ __launch_bounds__(256)
void k_scatter(const float* __restrict__ x, const int* __restrict__ ei,
               float* __restrict__ agg, float* __restrict__ cnt)
{
    const int t = blockIdx.x * 256 + threadIdx.x;
    const int eg = t >> 5;
    const int lane = t & 31;
    if (eg >= BB * EE) return;
    const int be = eg / EE, e = eg % EE;
    const int src = ei[e]      + be * NN;
    const int dst = ei[EE + e] + be * NN;
    const int f0 = lane * 4;
#pragma unroll
    for (int j = 0; j < 4; ++j)
        atomicAdd(&agg[(size_t)dst * HH + f0 + j], x[(size_t)src * HH + f0 + j]);
    if (lane == 0) atomicAdd(&cnt[dst], 1.0f);
}

__global__ __launch_bounds__(256)
void k_zero2(float* __restrict__ p1, int n1, float* __restrict__ p2, int n2)
{
    const int stride = gridDim.x * 256;
    for (int i = blockIdx.x * 256 + threadIdx.x; i < n1; i += stride) p1[i] = 0.0f;
    for (int i = blockIdx.x * 256 + threadIdx.x; i < n2; i += stride) p2[i] = 0.0f;
}

__global__ __launch_bounds__(256)
void k_head(const float* __restrict__ h, const float* __restrict__ Wo,
            const float* __restrict__ bo, float* __restrict__ out)
{
    const int r = blockIdx.x * 256 + threadIdx.x;
    if (r >= BN) return;
    float s = bo[0];
#pragma unroll 8
    for (int k = 0; k < HH; ++k) s += h[(size_t)r * HH + k] * Wo[k];
    out[r] = s;
}

// ---------------------------------------------------------------------------
extern "C" void kernel_launch(void* const* d_in, const int* in_sizes, int n_in,
                              void* d_out, int out_size, void* d_ws, size_t ws_size,
                              hipStream_t stream)
{
    const float* dyn   = (const float*)d_in[0];
    const float* sta   = (const float*)d_in[1];
    const int*   ei    = (const int*)  d_in[2];
    const float* W_ih  = (const float*)d_in[3];
    const float* W_hh  = (const float*)d_in[4];
    const float* b_ih  = (const float*)d_in[5];
    const float* b_hh  = (const float*)d_in[6];
    const float* W_sta = (const float*)d_in[7];
    const float* b_sta = (const float*)d_in[8];
    const float* W_fuse= (const float*)d_in[9];
    const float* b_fuse= (const float*)d_in[10];
    const float* s1Wl  = (const float*)d_in[11];
    const float* s1bl  = (const float*)d_in[12];
    const float* s1Wr  = (const float*)d_in[13];
    const float* s2Wl  = (const float*)d_in[14];
    const float* s2bl  = (const float*)d_in[15];
    const float* s2Wr  = (const float*)d_in[16];
    const float* W_out = (const float*)d_in[17];
    const float* b_out = (const float*)d_in[18];

    float* ws   = (float*)d_ws;
    float* z    = ws;                     // [BN,128]
    float* stil = z    + (size_t)BN * HH; // [BN,128]
    float* node = stil + (size_t)BN * HH; // [BN,128] (also reused as h2)
    float* agg  = node + (size_t)BN * HH; // [BN,128]
    float* h1   = agg  + (size_t)BN * HH; // [BN,128]
    float* cnt  = h1   + (size_t)BN * HH; // [BN]

    // 1) LSTM + static encoder
    k_lstm_static<<<BN / 16, 256, 0, stream>>>(dyn, sta, W_ih, W_hh, b_ih, b_hh,
                                               W_sta, b_sta, z, stil);
    // 2) fusion: node = relu([z|s] @ W_fuse^T + b)
    k_gemm256<<<BN / 64, 256, 0, stream>>>(z, stil, nullptr,
                                           W_fuse, 2 * HH, W_fuse + HH, 2 * HH,
                                           b_fuse, 1, node);
    // 3) SAGE layer 1
    k_zero2<<<512, 256, 0, stream>>>(agg, BN * HH, cnt, BN);
    k_scatter<<<(BB * EE * 32) / 256, 256, 0, stream>>>(node, ei, agg, cnt);
    k_gemm256<<<BN / 64, 256, 0, stream>>>(agg, node, cnt,
                                           s1Wl, HH, s1Wr, HH, s1bl, 1, h1);
    // 4) SAGE layer 2
    k_zero2<<<512, 256, 0, stream>>>(agg, BN * HH, cnt, BN);
    k_scatter<<<(BB * EE * 32) / 256, 256, 0, stream>>>(h1, ei, agg, cnt);
    k_gemm256<<<BN / 64, 256, 0, stream>>>(agg, h1, cnt,
                                           s2Wl, HH, s2Wr, HH, s2bl, 0, node);
    // 5) output head
    k_head<<<(BN + 255) / 256, 256, 0, stream>>>(node, W_out, b_out, (float*)d_out);
}
